// ResNet_29429115912467
// MI455X (gfx1250) — compile-verified, matrix operand order checked by reading, not run on hardware
//
#include <hip/hip_runtime.h>
#include <hip/hip_bf16.h>

// ---------------------------------------------------------------------------
// Types for CDNA5 WMMA (wave32): 16x16x32 bf16 -> f32
// ---------------------------------------------------------------------------
typedef __attribute__((ext_vector_type(16))) __bf16 v16bf;
typedef __attribute__((ext_vector_type(8)))  float  v8f;

union FragBF {
    uint4 q[2];
    v16bf v;
};

__device__ __forceinline__ unsigned short f2bf(float f) {
    unsigned int u = __float_as_uint(f);
    unsigned int r = u + 0x7fffu + ((u >> 16) & 1u);   // round-to-nearest-even
    return (unsigned short)(r >> 16);
}

#define KDEC_MAX 4608   // max K = 512*3*3

// ---- Tensor Data Mover availability ----------------------------------------
#if defined(__has_builtin)
#  if __has_builtin(__builtin_amdgcn_tensor_load_to_lds)
#    define HAVE_TDM 1
#  endif
#endif
#ifndef HAVE_TDM
#  define HAVE_TDM 0
#endif

#if HAVE_TDM
typedef __attribute__((ext_vector_type(4))) unsigned int u32x4;
typedef __attribute__((ext_vector_type(8))) int          i32x8;
typedef __attribute__((ext_vector_type(4))) int          i32x4;

// Issue a TDM 2D tile load: 64 rows x 64 bf16 cols from row-major [rows x K]
// bf16 matrix into LDS offset `ldsoff`. OOB columns (beyond cols_rem) read as
// zero -> implements our K-remainder zero padding in hardware.
__device__ __forceinline__ void tdm_load_tile(
    const unsigned short* gsrc, unsigned int ldsoff,
    unsigned int rows_rem, unsigned int cols_rem, unsigned int K)
{
    unsigned long long ga = (unsigned long long)gsrc;
    u32x4 g0;
    g0[0] = 1u;                                        // count=1, user mode
    g0[1] = ldsoff;                                    // lds_addr
    g0[2] = (unsigned int)ga;                          // global_addr[31:0]
    g0[3] = (unsigned int)(ga >> 32) | (2u << 30);     // global_addr[56:32] | type=2
    unsigned int w[8] = {0u, 0u, 0u, 0u, 0u, 0u, 0u, 0u};
    w[0]  = 1u << 16;                                  // data_size = 1 (2 bytes)
    w[1] |= (cols_rem & 0xFFFFu) << 16;                // tensor_dim0[15:0]  @bit48
    w[2] |= (cols_rem >> 16) | ((rows_rem & 0xFFFFu) << 16); // td0[31:16], td1[15:0]
    w[3] |= (rows_rem >> 16) | (64u << 16);            // td1[31:16], tile_dim0=64
    w[4]  = 64u;                                       // tile_dim1=64, tile_dim2=0
    w[5]  = K;                                         // tensor_dim0_stride[31:0]
    i32x8 g1;
#pragma unroll
    for (int i = 0; i < 8; ++i) g1[i] = (int)w[i];
    i32x4 z4 = {0, 0, 0, 0};
#if __clang_major__ >= 23
    i32x8 z8 = {0, 0, 0, 0, 0, 0, 0, 0};
    __builtin_amdgcn_tensor_load_to_lds(g0, g1, z4, z4, z8, 0);
#else
    __builtin_amdgcn_tensor_load_to_lds(g0, g1, z4, z4, 0);
#endif
}
#endif // HAVE_TDM

// Dynamic-LDS layout for conv kernel (dynamic LDS starts at offset 0: TDM dest)
#define SM_A      0                    // 64*64 bf16 = 8192 B  (TDM target)
#define SM_B      8192                 // 64*64 bf16 = 8192 B
#define SM_BOFF   16384                // 64 * int   = 256 B
#define SM_H0     16640                // 64 * short = 128 B
#define SM_W0     16768                // 64 * short = 128 B
#define SM_CI     16896                // 4608 * short = 9216 B
#define SM_KR     26112
#define SM_KS     35328
#define SM_BYTES  44544

// ---------------------------------------------------------------------------
// Implicit-GEMM conv (NCHW in, bf16 row-major weights, NCHW out), f32 accum.
// M = Cout (mult of 64), N = B*Ho*Wo (mult of 64), K = Cin*KH*KW (any).
// 256 threads = 8 waves; macro-tile 64(M) x 64(N), K-chunk 64.
// Wave owns 16(M) x 32(N) -> 2 accumulators, A-fragment reuse, 4 v_wmma/chunk.
// A-tile staged by the Tensor Data Mover (async, overlapped with the VALU
// im2col+convert staging of the B-tile); completion via s_wait_tensorcnt.
// ---------------------------------------------------------------------------
__global__ __launch_bounds__(256) void conv_wmma(
    const float* __restrict__ in, const unsigned short* __restrict__ wbf,
    float* __restrict__ out,
    int Bn, int Cin, int H, int W, int Cout, int Ho, int Wo,
    int KH, int KW, int stride, int pad)
{
    extern __shared__ __align__(16) unsigned char smem[];
    unsigned short* Alds  = (unsigned short*)(smem + SM_A);
    unsigned short* Blds  = (unsigned short*)(smem + SM_B);
    int*            s_boff= (int*)(smem + SM_BOFF);
    short*          s_h0  = (short*)(smem + SM_H0);
    short*          s_w0  = (short*)(smem + SM_W0);
    short*          s_ci  = (short*)(smem + SM_CI);
    short*          s_kr  = (short*)(smem + SM_KR);
    short*          s_ks  = (short*)(smem + SM_KS);

    const int K    = Cin * KH * KW;
    const int tid  = threadIdx.x;
    const int lane = tid & 31;
    const int wave = tid >> 5;
    const int m_sub = wave >> 1;         // 0..3 -> M offset 16*m_sub
    const int n_sub = wave & 1;          // 0..1 -> N offset 32*n_sub
    const int cout_base = blockIdx.x * 64;
    const int n_base    = blockIdx.y * 64;

    // ---- one-time per-block: n -> (b,ho,wo) decomposition (64 entries) ----
    if (tid < 64) {
        int n  = n_base + tid;
        int wo = n % Wo; int t = n / Wo; int ho = t % Ho; int b = t / Ho;
        s_boff[tid] = b * Cin * H * W;
        s_h0[tid]   = (short)(ho * stride - pad);
        s_w0[tid]   = (short)(wo * stride - pad);
    }
    // ---- one-time per-conv: k -> (ci,r,s) decomposition ----
    {
        const int khkw = KH * KW;
        for (int k = tid; k < K; k += 256) {
            int ci = k / khkw; int rs = k - ci * khkw;
            s_ci[k] = (short)ci;
            s_kr[k] = (short)(rs / KW);
            s_ks[k] = (short)(rs - (rs / KW) * KW);
        }
    }

    v8f acc0 = {}, acc1 = {};

    const int kbase = (lane >> 4) << 3;            // 0 or 8 (ISA 16-bit A/B layout)
    const int rowA  = m_sub * 16 + (lane & 15);
    const int rowB0 = n_sub * 32 + (lane & 15);
    const int rowB1 = rowB0 + 16;

    for (int k0 = 0; k0 < K; k0 += 64) {
        __syncthreads();   // covers one-time tables on iter 0; tile reuse after

#if HAVE_TDM
        // Stage A via Tensor Data Mover (wave 0 issues one 2D-tile descriptor)
        if (wave == 0) {
            tdm_load_tile(&wbf[(size_t)cout_base * K + k0], (unsigned int)SM_A,
                          (unsigned int)(Cout - cout_base),
                          (unsigned int)(K - k0), (unsigned int)K);
        }
#else
        // Fallback: stage A with vector loads/ds stores (zero-pad K remainder)
        for (int i = tid; i < 64 * 64; i += 256) {
            int m = i >> 6, kk = i & 63, k = k0 + kk;
            Alds[i] = (k < K) ? wbf[(size_t)(cout_base + m) * K + k]
                              : (unsigned short)0;
        }
#endif
        // Stage B: im2col patches, 64(n) x 64(k) bf16 — division-free
        for (int i = tid; i < 64 * 64; i += 256) {
            int nl = i >> 6, kk = i & 63, k = k0 + kk;
            float v = 0.f;
            if (k < K) {
                int h = s_h0[nl] + s_kr[k];
                int w = s_w0[nl] + s_ks[k];
                if (h >= 0 && h < H && w >= 0 && w < W)
                    v = in[(size_t)s_boff[nl] + ((size_t)s_ci[k] * H + h) * W + w];
            }
            Blds[(nl << 6) + kk] = f2bf(v);
        }
#if HAVE_TDM
        __builtin_amdgcn_s_wait_tensorcnt(0);   // TENSORcnt==0 (non-issuers pass)
#endif
        __syncthreads();

        // prefetch next weight chunk into cache (global_prefetch_b8)
        if (k0 + 64 < K)
            __builtin_prefetch(&wbf[(size_t)(cout_base + (tid >> 2)) * K + k0 + 64], 0, 1);

        // 2 K-subchunks x 2 N-subtiles = 4 v_wmma, A fragment reused
#pragma unroll
        for (int s = 0; s < 2; ++s) {
            const int koff = s * 32 + kbase;
            FragBF fa, fb0, fb1;
            fa.q[0]  = *(const uint4*)&Alds[(rowA  << 6) + koff];
            fa.q[1]  = *(const uint4*)&Alds[(rowA  << 6) + 16 + koff];
            fb0.q[0] = *(const uint4*)&Blds[(rowB0 << 6) + koff];
            fb0.q[1] = *(const uint4*)&Blds[(rowB0 << 6) + 16 + koff];
            fb1.q[0] = *(const uint4*)&Blds[(rowB1 << 6) + koff];
            fb1.q[1] = *(const uint4*)&Blds[(rowB1 << 6) + 16 + koff];
            acc0 = __builtin_amdgcn_wmma_f32_16x16x32_bf16(
                false, fa.v, false, fb0.v, (short)0, acc0, false, false);
            acc1 = __builtin_amdgcn_wmma_f32_16x16x32_bf16(
                false, fa.v, false, fb1.v, (short)0, acc1, false, false);
        }
    }

    // D writeback per ISA 32-bit C/D layout: VGPR r -> M=r+8*(lane>=16); N=lane&15
    const int moff = cout_base + m_sub * 16 + ((lane >> 4) << 3);
#pragma unroll
    for (int t = 0; t < 2; ++t) {
        const int n  = n_base + n_sub * 32 + t * 16 + (lane & 15);
        const int wo = n % Wo; const int tt = n / Wo;
        const int ho = tt % Ho; const int b = tt / Ho;
        const v8f& a = t ? acc1 : acc0;
#pragma unroll
        for (int r = 0; r < 8; ++r) {
            out[(((size_t)b * Cout + (moff + r)) * Ho + ho) * Wo + wo] = a[r];
        }
    }
}

// ---------------------------------------------------------------------------
// Per-layer fp32 -> bf16 weight conversion (row-major, element-wise)
// ---------------------------------------------------------------------------
__global__ void w_to_bf16(const float* __restrict__ w,
                          unsigned short* __restrict__ o, int n)
{
    int i = blockIdx.x * 256 + threadIdx.x;
    if (i < n) o[i] = f2bf(w[i]);
}

// ---------------------------------------------------------------------------
// BN batch statistics: one block per channel -> stats[2c]=mean, stats[2c+1]=invstd
// ---------------------------------------------------------------------------
__global__ __launch_bounds__(256) void bn_stats(
    const float* __restrict__ x, float* __restrict__ stats,
    int Bn, int C, int HW)
{
    const int c = blockIdx.x;
    float s = 0.f, s2 = 0.f;
    for (int b = 0; b < Bn; ++b) {
        const float* p = x + ((size_t)b * C + c) * HW;
        for (int i = threadIdx.x; i < HW; i += 256) {
            float v = p[i]; s += v; s2 += v * v;
        }
    }
    __shared__ float r1[256], r2[256];
    r1[threadIdx.x] = s; r2[threadIdx.x] = s2;
    __syncthreads();
    for (int o = 128; o > 0; o >>= 1) {
        if (threadIdx.x < o) {
            r1[threadIdx.x] += r1[threadIdx.x + o];
            r2[threadIdx.x] += r2[threadIdx.x + o];
        }
        __syncthreads();
    }
    if (threadIdx.x == 0) {
        float n   = (float)Bn * (float)HW;
        float m   = r1[0] / n;
        float var = r2[0] / n - m * m;
        stats[2 * c]     = m;
        stats[2 * c + 1] = rsqrtf(var + 1e-5f);
    }
}

// ---------------------------------------------------------------------------
// Fused BN-apply (+ optional residual, + optional ReLU); in-place safe.
// ---------------------------------------------------------------------------
__global__ void bn_apply(
    const float* __restrict__ x, float* __restrict__ y,
    const float* __restrict__ stats, const float* __restrict__ g,
    const float* __restrict__ bt, const float* __restrict__ res,
    int C, int HW, size_t total, int relu)
{
    size_t i = (size_t)blockIdx.x * 256 + threadIdx.x;
    if (i >= total) return;
    int c = (int)((i / (size_t)HW) % (size_t)C);
    float v = (x[i] - stats[2 * c]) * stats[2 * c + 1] * g[c] + bt[c];
    if (res) v += res[i];
    if (relu) v = fmaxf(v, 0.f);
    y[i] = v;
}

// ---------------------------------------------------------------------------
// 3x3 stride-2 pad-1 max pool
// ---------------------------------------------------------------------------
__global__ void maxpool3x3s2(
    const float* __restrict__ in, float* __restrict__ out,
    int Bn, int C, int H, int W, int OH, int OW)
{
    size_t i = (size_t)blockIdx.x * 256 + threadIdx.x;
    size_t total = (size_t)Bn * C * OH * OW;
    if (i >= total) return;
    int ow = (int)(i % OW); size_t t = i / OW;
    int oh = (int)(t % OH); t /= OH;
    int c  = (int)(t % C);  int b = (int)(t / C);
    const float* p = in + ((size_t)b * C + c) * H * W;
    float m = -3.402823466e38f;
    for (int r = 0; r < 3; ++r) {
        int h = oh * 2 - 1 + r;
        if (h < 0 || h >= H) continue;
        for (int s = 0; s < 3; ++s) {
            int w = ow * 2 - 1 + s;
            if (w < 0 || w >= W) continue;
            m = fmaxf(m, p[h * W + w]);
        }
    }
    out[i] = m;
}

// ---------------------------------------------------------------------------
// Per-sample 2-center k-means on [64, 64x48] feature map -> binary mask.
// ---------------------------------------------------------------------------
__global__ __launch_bounds__(256) void kmeans_mask_k(
    const float* __restrict__ xp, float* __restrict__ mask)
{
    const int b = blockIdx.x;
    const float* X = xp + (size_t)b * 64 * 3072;   // X[c*3072 + n]
    __shared__ float c0[64], c1[64];
    __shared__ unsigned char asg[3072];
    __shared__ int cnt1;
    const int tid = threadIdx.x;

    if (tid < 64) {
        c0[tid] = X[tid * 3072 + 0];
        c1[tid] = X[tid * 3072 + 1536];
    }
    __syncthreads();

    for (int it = 0; it < 11; ++it) {              // 10 updates + final assignment
        for (int n = tid; n < 3072; n += 256) {
            float d0 = 0.f, d1 = 0.f;
            for (int c = 0; c < 64; ++c) {
                float v = X[c * 3072 + n];
                float a = v - c0[c]; float e = v - c1[c];
                d0 += a * a; d1 += e * e;
            }
            asg[n] = (d1 < d0) ? 1 : 0;            // argmin; tie -> cluster 0
        }
        if (tid == 0) cnt1 = 0;
        __syncthreads();
        if (it == 10) break;                       // uniform exit

        int lc = 0;
        for (int n = tid; n < 3072; n += 256) lc += asg[n];
        atomicAdd(&cnt1, lc);
        __syncthreads();
        float n1 = (float)cnt1, n0 = 3072.f - n1;
        if (tid < 128) {
            int k = tid >> 6, c = tid & 63;
            float s = 0.f;
            for (int n = 0; n < 3072; ++n)
                if ((int)asg[n] == k) s += X[c * 3072 + n];
            float denom = fmaxf(k ? n1 : n0, 1.f);
            if (k) c1[c] = s / denom; else c0[c] = s / denom;
        }
        __syncthreads();
    }
    __syncthreads();
    const unsigned char ref = asg[32 * 48 + 24];   // idx[32,24]
    for (int n = tid; n < 3072; n += 256)
        mask[(size_t)b * 3072 + n] = (asg[n] == ref) ? 1.f : 0.f;
}

// ---------------------------------------------------------------------------
// Bilinear resize (half-pixel centers, edge clamp) of [B,1,ih,iw] -> [B,1,oh,ow]
// ---------------------------------------------------------------------------
__global__ void resize_bilinear(
    const float* __restrict__ in, float* __restrict__ out,
    int Bn, int ih, int iw, int oh, int ow)
{
    size_t i = (size_t)blockIdx.x * 256 + threadIdx.x;
    size_t total = (size_t)Bn * oh * ow;
    if (i >= total) return;
    int x = (int)(i % ow); size_t t = i / ow;
    int y = (int)(t % oh); int b = (int)(t / oh);
    float sy = (y + 0.5f) * (float)ih / (float)oh - 0.5f;
    float sx = (x + 0.5f) * (float)iw / (float)ow - 0.5f;
    int y0 = (int)floorf(sy); float fy = sy - (float)y0;
    int x0 = (int)floorf(sx); float fx = sx - (float)x0;
    int y0c = min(max(y0, 0), ih - 1), y1c = min(max(y0 + 1, 0), ih - 1);
    int x0c = min(max(x0, 0), iw - 1), x1c = min(max(x0 + 1, 0), iw - 1);
    const float* p = in + (size_t)b * ih * iw;
    float v = p[y0c * iw + x0c] * (1.f - fy) * (1.f - fx)
            + p[y0c * iw + x1c] * (1.f - fy) * fx
            + p[y1c * iw + x0c] * fy * (1.f - fx)
            + p[y1c * iw + x1c] * fy * fx;
    out[i] = v;
}

// ---------------------------------------------------------------------------
// o = o * mask + o = o * (1 + mask)   (mask broadcast over channels)
// ---------------------------------------------------------------------------
__global__ void mask_scale(
    float* __restrict__ o, const float* __restrict__ m,
    int C, int HW, size_t total)
{
    size_t i = (size_t)blockIdx.x * 256 + threadIdx.x;
    if (i >= total) return;
    size_t hw = i % (size_t)HW;
    size_t b  = i / ((size_t)C * HW);
    o[i] *= (1.f + m[b * HW + hw]);
}

// ---------------------------------------------------------------------------
// Pack outputs: (o4, o4, mask) concatenated flat
// ---------------------------------------------------------------------------
__global__ void pack_out(
    const float* __restrict__ o4, const float* __restrict__ mask,
    float* __restrict__ out, int o4n, int mn)
{
    int i = blockIdx.x * 256 + threadIdx.x;
    if (i < o4n) { out[i] = o4[i]; out[o4n + i] = o4[i]; }
    if (i < mn)  out[2 * o4n + i] = mask[i];
}

// ---------------------------------------------------------------------------
// Host orchestration
// ---------------------------------------------------------------------------
extern "C" void kernel_launch(void* const* d_in, const int* in_sizes, int n_in,
                              void* d_out, int out_size, void* d_ws, size_t ws_size,
                              hipStream_t stream) {
    (void)out_size; (void)ws_size;
    const int Bn = 32;

    // ---- locate x (uniquely 32*3*256*192 elements) and collect param leaves ----
    const long long XSZ = 32ll * 3 * 256 * 192;
    int xi = 0;
    for (int i = 0; i < n_in; ++i)
        if ((long long)in_sizes[i] == XSZ) { xi = i; break; }
    const float* x = (const float*)d_in[xi];

    const float* P[80]; int PS[80]; int np = 0;
    for (int i = 0; i < n_in; ++i)
        if (i != xi) { P[np] = (const float*)d_in[i]; PS[np] = in_sizes[i]; ++np; }

    struct Blk { const float *w1,*g1,*b1,*w2,*g2,*b2,*wd,*gd,*bd; } blk[4][2];
    const float *conv1w, *g0, *b0;
    // sorted-key pytree order starts: b0(64), conv1(9408), g0(64);
    // insertion order starts: conv1(9408), g0(64), b0(64)
    bool sortedOrder = (np > 1 && PS[0] == 64 && PS[1] == 9408);
    int c = 0;
    if (sortedOrder) { b0 = P[0]; conv1w = P[1]; g0 = P[2]; c = 3; }
    else             { conv1w = P[0]; g0 = P[1]; b0 = P[2]; c = 3; }
    for (int si = 0; si < 4; ++si)
        for (int bi = 0; bi < 2; ++bi) {
            bool hasd = (si > 0 && bi == 0);
            Blk& q = blk[si][bi];
            q.wd = q.gd = q.bd = nullptr;
            if (sortedOrder) {
                q.b1 = P[c++]; q.b2 = P[c++]; if (hasd) q.bd = P[c++];
                q.g1 = P[c++]; q.g2 = P[c++]; if (hasd) q.gd = P[c++];
                q.w1 = P[c++]; q.w2 = P[c++]; if (hasd) q.wd = P[c++];
            } else {
                q.w1 = P[c++]; q.g1 = P[c++]; q.b1 = P[c++];
                q.w2 = P[c++]; q.g2 = P[c++]; q.b2 = P[c++];
                if (hasd) { q.wd = P[c++]; q.gd = P[c++]; q.bd = P[c++]; }
            }
        }

    // ---- workspace carve (floats). conv1buf (25.2M) overlaps bufA..D (dead after pool) ----
    float* ws = (float*)d_ws;
    const size_t ACT = 6291456;                 // 32*64*64*48 (max stage activation)
    float* bufA     = ws;
    float* bufB     = ws + ACT;
    float* bufC     = ws + 2 * ACT;
    float* bufD     = ws + 3 * ACT;
    float* conv1buf = ws;                       // 32*64*128*96 = 25,165,824 = 4*ACT
    float* poolbuf  = ws + 4 * ACT;
    float* mask64   = poolbuf + ACT;            // 32*64*48 = 98,304
    float* maskr    = mask64 + 98304;
    float* statsA   = maskr + 98304;            // 2*512 max
    float* statsB   = statsA + 1024;
    unsigned short* wbf = (unsigned short*)(statsB + 1024); // 2,359,296 bf16 max

    auto conv = [&](const float* in, const float* w, float* out,
                    int Cin, int H, int W, int Cout, int Ho, int Wo,
                    int KH, int KW, int st, int pad) {
        int K = Cin * KH * KW;
        int nw = Cout * K;
        w_to_bf16<<<(nw + 255) / 256, 256, 0, stream>>>(w, wbf, nw);
        dim3 g((unsigned)(Cout / 64), (unsigned)((Bn * Ho * Wo) / 64));
        conv_wmma<<<g, 256, SM_BYTES, stream>>>(in, wbf, out, Bn, Cin, H, W,
                                                Cout, Ho, Wo, KH, KW, st, pad);
    };
    auto stats = [&](const float* t, float* st, int C, int HW) {
        bn_stats<<<C, 256, 0, stream>>>(t, st, Bn, C, HW);
    };
    auto apply = [&](const float* xx, float* yy, const float* st, const float* g,
                     const float* bb, const float* res, int C, int HW, int relu) {
        size_t tot = (size_t)Bn * C * HW;
        bn_apply<<<(unsigned)((tot + 255) / 256), 256, 0, stream>>>(
            xx, yy, st, g, bb, res, C, HW, tot, relu);
    };

    // ---- stem: conv1 7x7 s2 p3 -> BN -> ReLU -> maxpool 3x3 s2 p1 ----
    conv(x, conv1w, conv1buf, 3, 256, 192, 64, 128, 96, 7, 7, 2, 3);
    stats(conv1buf, statsA, 64, 128 * 96);
    apply(conv1buf, conv1buf, statsA, g0, b0, nullptr, 64, 128 * 96, 1);
    {
        size_t tot = (size_t)Bn * 64 * 64 * 48;
        maxpool3x3s2<<<(unsigned)((tot + 255) / 256), 256, 0, stream>>>(
            conv1buf, poolbuf, Bn, 64, 128, 96, 64, 48);
    }

    // ---- k-means mask on pooled features ----
    kmeans_mask_k<<<Bn, 256, 0, stream>>>(poolbuf, mask64);

    // ---- residual stages ----
    const int widths[4] = {64, 128, 256, 512};
    const int hs[5] = {64, 64, 32, 16, 8};
    const int wz[5] = {48, 48, 24, 12, 6};
    const float* cur = poolbuf;

    for (int si = 0; si < 4; ++si) {
        const int Cin0 = (si == 0) ? 64 : widths[si - 1];
        const int Cw   = widths[si];
        const int Hin  = hs[si],     Win = wz[si];
        const int Ho   = hs[si + 1], Wo  = wz[si + 1];
        const int st0  = (si == 0) ? 1 : 2;
        const int HW   = Ho * Wo;

        // block0: in=cur, t=bufB, out=bufA, idn=bufD
        {
            Blk& q = blk[si][0];
            conv(cur, q.w1, bufB, Cin0, Hin, Win, Cw, Ho, Wo, 3, 3, st0, 1);
            stats(bufB, statsA, Cw, HW);
            apply(bufB, bufB, statsA, q.g1, q.b1, nullptr, Cw, HW, 1);
            conv(bufB, q.w2, bufA, Cw, Ho, Wo, Cw, Ho, Wo, 3, 3, 1, 1);
            stats(bufA, statsA, Cw, HW);
            const float* res = cur;
            if (q.wd) {
                conv(cur, q.wd, bufD, Cin0, Hin, Win, Cw, Ho, Wo, 1, 1, st0, 0);
                stats(bufD, statsB, Cw, HW);
                apply(bufD, bufD, statsB, q.gd, q.bd, nullptr, Cw, HW, 0);
                res = bufD;
            }
            apply(bufA, bufA, statsA, q.g2, q.b2, res, Cw, HW, 1);
        }
        // block1: in=bufA, t=bufB, out=bufC
        {
            Blk& q = blk[si][1];
            conv(bufA, q.w1, bufB, Cw, Ho, Wo, Cw, Ho, Wo, 3, 3, 1, 1);
            stats(bufB, statsA, Cw, HW);
            apply(bufB, bufB, statsA, q.g1, q.b1, nullptr, Cw, HW, 1);
            conv(bufB, q.w2, bufC, Cw, Ho, Wo, Cw, Ho, Wo, 3, 3, 1, 1);
            stats(bufC, statsA, Cw, HW);
            apply(bufC, bufC, statsA, q.g2, q.b2, bufA, Cw, HW, 1);
        }
        // stage mask: o = o * rs(mask) + o
        const float* m = mask64;
        if (si > 0) {
            size_t rt = (size_t)Bn * HW;
            resize_bilinear<<<(unsigned)((rt + 255) / 256), 256, 0, stream>>>(
                mask64, maskr, Bn, 64, 48, Ho, Wo);
            m = maskr;
        }
        {
            size_t tot = (size_t)Bn * Cw * HW;
            mask_scale<<<(unsigned)((tot + 255) / 256), 256, 0, stream>>>(
                bufC, m, Cw, HW, tot);
        }
        cur = bufC;
    }

    // ---- pack (o4, o4, mask) ----
    const int o4n = 32 * 512 * 8 * 6;    // 786,432
    const int mn  = 32 * 64 * 48;        // 98,304
    pack_out<<<(o4n + 255) / 256, 256, 0, stream>>>(bufC, mask64, (float*)d_out, o4n, mn);
}
